// MultiHeadAttention_79499844649593
// MI455X (gfx1250) — matrix operand order repaired and checked by reading, not compiled, not run on hardware
//
#include <hip/hip_runtime.h>
#include <hip/hip_bf16.h>

typedef __attribute__((ext_vector_type(16))) __bf16 v16bf;
typedef __attribute__((ext_vector_type(8)))  __bf16 v8bf;
typedef __attribute__((ext_vector_type(4)))  __bf16 v4bf;
typedef __attribute__((ext_vector_type(8)))  float  v8f;
typedef __attribute__((ext_vector_type(4)))  int    v4i;

#define D_MODEL 1024
#define SEQ     2048
#define NHEADS  16
#define HDIM    64
#define BATCH   4
#define MTOT    (BATCH * SEQ)   // 8192

// -------- async global->LDS path (guarded; falls back to sync staging) ------
#if defined(__has_builtin)
#  if __has_builtin(__builtin_amdgcn_global_load_async_to_lds_b128) && \
      __has_builtin(__builtin_amdgcn_s_wait_asynccnt)
#    define USE_ASYNC_LDS 1
#  endif
#endif
#ifndef USE_ASYNC_LDS
#  define USE_ASYNC_LDS 0
#endif
#define AS1 __attribute__((address_space(1)))
#define AS3 __attribute__((address_space(3)))

// ---------------------------------------------------------------------------
// Fragment loader: two 16-byte chunks -> one 16-element bf16 WMMA operand
// ---------------------------------------------------------------------------
__device__ __forceinline__ v16bf load_frag(const __bf16* p0, const __bf16* p1) {
    union { v16bf v; v8bf h[2]; } u;
    u.h[0] = *(const v8bf*)p0;
    u.h[1] = *(const v8bf*)p1;
    return u.v;
}

__device__ __forceinline__ float rowmax16(float v) {
    #pragma unroll
    for (int off = 1; off < 16; off <<= 1)
        v = fmaxf(v, __shfl_xor(v, off, 32));
    return v;
}
__device__ __forceinline__ float rowsum16(float v) {
    #pragma unroll
    for (int off = 1; off < 16; off <<= 1)
        v += __shfl_xor(v, off, 32);
    return v;
}

// ---------------------------------------------------------------------------
// fp32 -> bf16 conversion (4 elems / thread)
// ---------------------------------------------------------------------------
__global__ __launch_bounds__(256) void cvt4_kernel(const float4* __restrict__ in,
                                                   v4bf* __restrict__ out, int n4) {
    int i = blockIdx.x * 256 + threadIdx.x;
    if (i < n4) {
        float4 f = in[i];
        v4bf o;
        o[0] = (__bf16)f.x; o[1] = (__bf16)f.y;
        o[2] = (__bf16)f.z; o[3] = (__bf16)f.w;
        out[i] = o;
    }
}

// Transpose 1024x1024 fp32 -> bf16  (out[n*1024+k] = in[k*1024+n])
__global__ __launch_bounds__(256) void transpose_w_kernel(const float* __restrict__ in,
                                                          __bf16* __restrict__ out) {
    int idx = blockIdx.x * 256 + threadIdx.x;
    int k = idx & 1023;
    int n = idx >> 10;
    out[(size_t)idx] = (__bf16)in[(size_t)k * D_MODEL + n];
}

// ---------------------------------------------------------------------------
// bf16 WMMA GEMM:  C[M=8192, N=1024] = A[M,1024] @ Bt[N,1024]^T
//   block = 256 threads (8 waves, 4(M) x 2(N)); block tile 128x128
//   wave tile 32x64 = 2x4 WMMA accumulators; K-step 32
//   A/B K-slices double-buffered in LDS, filled via async global->LDS copies
// MODE 0: fp32 out + bias          (output projection)
// MODE 1: bf16 out, [M,N] layout   (Q, K)
// MODE 2: bf16 out -> Vt[b,h,hd,S] (V, transposed for attention)
// ---------------------------------------------------------------------------
template <int MODE>
__global__ __launch_bounds__(256) void gemm_bf16_kernel(const __bf16* __restrict__ A,
                                                        const __bf16* __restrict__ Bt,
                                                        void* __restrict__ outp,
                                                        const float* __restrict__ bias) {
    const int Kd = D_MODEL, Nd = D_MODEL;
    const int tid  = threadIdx.x;
    const int lane = tid & 31;
    const int w    = tid >> 5;
    const int g    = lane >> 4;     // half-group 0/1
    const int lq   = lane & 15;
    const int wm   = w >> 1;        // 0..3  (M)
    const int wn   = w & 1;         // 0..1  (N)

    const int m0b = blockIdx.x * 128;
    const int n0b = blockIdx.y * 128;

    __shared__ __bf16 Ab[2][128][32];   // 8 KB x2
    __shared__ __bf16 Bb[2][128][32];   // 8 KB x2

    // staging: each thread copies one 32B run (two b128) per tile per K-step
    const int srow  = tid >> 1;          // 0..127
    const int spart = (tid & 1) * 16;    // element offset {0,16} in the 32-wide slice
    const __bf16* gA = A  + (size_t)(m0b + srow) * Kd + spart;
    const __bf16* gB = Bt + (size_t)(n0b + srow) * Kd + spart;

    auto stage = [&](int buf, int kk) {
#if USE_ASYNC_LDS
        AS1 v4i* ga = (AS1 v4i*)(gA + kk);
        AS1 v4i* gb = (AS1 v4i*)(gB + kk);
        AS3 v4i* la = (AS3 v4i*)&Ab[buf][srow][spart];
        AS3 v4i* lb = (AS3 v4i*)&Bb[buf][srow][spart];
        __builtin_amdgcn_global_load_async_to_lds_b128(ga, la, 0, 0);
        __builtin_amdgcn_global_load_async_to_lds_b128(ga, la, 16, 0);
        __builtin_amdgcn_global_load_async_to_lds_b128(gb, lb, 0, 0);
        __builtin_amdgcn_global_load_async_to_lds_b128(gb, lb, 16, 0);
#else
        *(v8bf*)&Ab[buf][srow][spart]     = *(const v8bf*)(gA + kk);
        *(v8bf*)&Ab[buf][srow][spart + 8] = *(const v8bf*)(gA + kk + 8);
        *(v8bf*)&Bb[buf][srow][spart]     = *(const v8bf*)(gB + kk);
        *(v8bf*)&Bb[buf][srow][spart + 8] = *(const v8bf*)(gB + kk + 8);
#endif
    };

    v8f acc[2][4] = {};
    const int ka = g * 16;   // A-fragment K sub-offset (contiguous, matches B)
    const int kb = g * 16;   // B-fragment K sub-offset (contiguous, matches A)
    const int ar0 = wm * 32 + lq;        // local A rows for the two M-tiles
    const int ar1 = wm * 32 + 16 + lq;

    stage(0, 0);
#if USE_ASYNC_LDS
    __builtin_amdgcn_s_wait_asynccnt(0);
#endif
    __syncthreads();

    int cur = 0;
    for (int kk = 0; kk < Kd; kk += 32) {
        const int nxt = kk + 32;
        if (nxt < Kd) stage(cur ^ 1, nxt);     // prefetch next slice (async)

        v16bf a0 = load_frag(&Ab[cur][ar0][ka], &Ab[cur][ar0][ka + 8]);
        v16bf a1 = load_frag(&Ab[cur][ar1][ka], &Ab[cur][ar1][ka + 8]);
        #pragma unroll
        for (int j = 0; j < 4; ++j) {
            const int br = wn * 64 + j * 16 + lq;
            v16bf bf = load_frag(&Bb[cur][br][kb], &Bb[cur][br][kb + 8]);
            acc[0][j] = __builtin_amdgcn_wmma_f32_16x16x32_bf16(false, a0, false, bf, (short)0, acc[0][j], false, false);
            acc[1][j] = __builtin_amdgcn_wmma_f32_16x16x32_bf16(false, a1, false, bf, (short)0, acc[1][j], false, false);
        }

        if (nxt < Kd) {
#if USE_ASYNC_LDS
            __builtin_amdgcn_s_wait_asynccnt(0);
#endif
            __syncthreads();
            cur ^= 1;
        }
    }

    #pragma unroll
    for (int i = 0; i < 2; ++i) {
        #pragma unroll
        for (int j = 0; j < 4; ++j) {
            const int tm = m0b + wm * 32 + i * 16;
            const int tn = n0b + wn * 64 + j * 16;
            #pragma unroll
            for (int r = 0; r < 8; ++r) {
                const int row = tm + r + g * 8;       // ISA f32 C/D layout
                const int col = tn + lq;
                const float v = acc[i][j][r];
                if (MODE == 0) {
                    ((float*)outp)[(size_t)row * Nd + col] = v + bias[col];
                } else if (MODE == 1) {
                    ((__bf16*)outp)[(size_t)row * Nd + col] = (__bf16)v;
                } else {
                    const int b = row >> 11, s = row & (SEQ - 1);
                    const int h = col >> 6,  d = col & (HDIM - 1);
                    ((__bf16*)outp)[((size_t)((b * NHEADS + h) * HDIM + d)) * SEQ + s] = (__bf16)v;
                }
            }
        }
    }
}

// ---------------------------------------------------------------------------
// Causal flash attention. One wave per (b, h, 16 query rows).
// Q,K: bf16 [B,S,H*hd]; Vt: bf16 [B,H,hd,S]; ctx out: bf16 [B,S,H*hd]
// Key step = 32 (one full WMMA K-dim for P@V).
// ---------------------------------------------------------------------------
__global__ __launch_bounds__(32) void attn_kernel(const __bf16* __restrict__ Q,
                                                  const __bf16* __restrict__ Kb,
                                                  const __bf16* __restrict__ Vt,
                                                  __bf16* __restrict__ ctx) {
    const int lane = threadIdx.x;
    const int g  = lane >> 4;
    const int lq = lane & 15;
    const int b  = blockIdx.z;
    const int h  = blockIdx.y;
    const int m0 = blockIdx.x * 16;
    const int mlast = m0 + 15;

    const __bf16* qbase = Q  + ((size_t)(b * SEQ + m0)) * D_MODEL + h * HDIM;
    const __bf16* kbase = Kb + ((size_t)(b * SEQ))      * D_MODEL + h * HDIM;
    const __bf16* vbase = Vt + ((size_t)((b * NHEADS + h) * HDIM)) * SEQ;

    // Q A-fragments: hd chunks [0,32) and [32,64)
    const __bf16* qrow = qbase + (size_t)lq * D_MODEL;
    const int ka = g * 16;
    v16bf qa0 = load_frag(qrow + 0  + ka, qrow + 0  + ka + 8);
    v16bf qa1 = load_frag(qrow + 32 + ka, qrow + 32 + ka + 8);

    v8f c[4] = {};                 // 16x64 ctx accumulator
    float mr[8], lr[8];
    #pragma unroll
    for (int r = 0; r < 8; ++r) { mr[r] = -3.0e38f; lr[r] = 0.0f; }

    __shared__ __bf16 pl[16][32];  // P staging tile

    const float sc = 0.125f * 1.44269504088896f;   // 1/sqrt(64) * log2(e)

    for (int j0 = 0; j0 <= mlast; j0 += 32) {
        // ---- scores: S(16x32) = Q(16x64) @ K^T(64x32), 4 WMMAs ----
        v8f s[2];
        #pragma unroll
        for (int nt = 0; nt < 2; ++nt) {
            int key  = j0 + nt * 16 + lq;
            int kidx = key < SEQ ? key : SEQ - 1;          // clamp (masked below)
            const __bf16* krow = kbase + (size_t)kidx * D_MODEL;
            v16bf kb0 = load_frag(krow + 0  + g * 16, krow + 0  + g * 16 + 8);
            v16bf kb1 = load_frag(krow + 32 + g * 16, krow + 32 + g * 16 + 8);
            v8f z = {};
            z = __builtin_amdgcn_wmma_f32_16x16x32_bf16(false, qa0, false, kb0, (short)0, z, false, false);
            z = __builtin_amdgcn_wmma_f32_16x16x32_bf16(false, qa1, false, kb1, (short)0, z, false, false);
            s[nt] = z;
        }

        // ---- online softmax (exp2 domain), per-row stats via half-group shuffles ----
        float p0[8], p1[8];
        #pragma unroll
        for (int r = 0; r < 8; ++r) {
            const int qr  = m0 + r + g * 8;
            const int k0i = j0 + lq, k1i = j0 + 16 + lq;
            float v0 = (k0i <= qr) ? s[0][r] * sc : -3.0e38f;
            float v1 = (k1i <= qr) ? s[1][r] * sc : -3.0e38f;
            float rm   = rowmax16(fmaxf(v0, v1));
            float mnew = fmaxf(mr[r], rm);
            float alpha = exp2f(mr[r] - mnew);
            float e0 = exp2f(v0 - mnew);
            float e1 = exp2f(v1 - mnew);
            lr[r] = lr[r] * alpha + rowsum16(e0 + e1);
            mr[r] = mnew;
            p0[r] = e0; p1[r] = e1;
            c[0][r] *= alpha; c[1][r] *= alpha; c[2][r] *= alpha; c[3][r] *= alpha;
        }

        // ---- P (C-layout) -> LDS -> A-fragment ----
        #pragma unroll
        for (int r = 0; r < 8; ++r) {
            pl[r + g * 8][lq]      = (__bf16)p0[r];
            pl[r + g * 8][16 + lq] = (__bf16)p1[r];
        }
        __syncthreads();
        v16bf pa = load_frag(&pl[lq][ka], &pl[lq][ka + 8]);
        __syncthreads();

        // ---- ctx += P(16x32) @ V(32x64), 4 WMMAs; Vt contiguous along S ----
        #pragma unroll
        for (int nt2 = 0; nt2 < 4; ++nt2) {
            const int d = nt2 * 16 + lq;
            const __bf16* vrow = vbase + (size_t)d * SEQ + j0 + g * 16;
            v16bf vb = load_frag(vrow, vrow + 8);
            c[nt2] = __builtin_amdgcn_wmma_f32_16x16x32_bf16(false, pa, false, vb, (short)0, c[nt2], false, false);
        }
    }

    // ---- normalize + store bf16 ctx in [B,S,H*hd] ----
    #pragma unroll
    for (int r = 0; r < 8; ++r) {
        const int srow = m0 + r + g * 8;
        __bf16* orow = ctx + ((size_t)(b * SEQ + srow)) * D_MODEL + h * HDIM;
        const float invl = 1.0f / lr[r];
        #pragma unroll
        for (int nt2 = 0; nt2 < 4; ++nt2)
            orow[nt2 * 16 + lq] = (__bf16)(c[nt2][r] * invl);
    }
}

// ---------------------------------------------------------------------------
// Launcher
// ---------------------------------------------------------------------------
extern "C" void kernel_launch(void* const* d_in, const int* in_sizes, int n_in,
                              void* d_out, int out_size, void* d_ws, size_t ws_size,
                              hipStream_t stream) {
    const float* x  = (const float*)d_in[0];
    const float* Wq = (const float*)d_in[1];
    const float* Wk = (const float*)d_in[2];
    const float* Wv = (const float*)d_in[3];
    const float* Wo = (const float*)d_in[4];
    const float* bo = (const float*)d_in[5];
    float* out = (float*)d_out;

    const size_t MB = 1024u * 1024u;
    char* ws = (char*)d_ws;
    __bf16* xbf = (__bf16*)(ws + 0 * MB);   // 16 MB
    __bf16* wqT = (__bf16*)(ws + 16 * MB);  //  2 MB each
    __bf16* wkT = (__bf16*)(ws + 18 * MB);
    __bf16* wvT = (__bf16*)(ws + 20 * MB);
    __bf16* woT = (__bf16*)(ws + 22 * MB);
    __bf16* Qb  = (__bf16*)(ws + 24 * MB);  // 16 MB
    __bf16* Kbb = (__bf16*)(ws + 40 * MB);  // 16 MB
    __bf16* Vt  = (__bf16*)(ws + 56 * MB);  // 16 MB
    __bf16* Cb  = (__bf16*)(ws + 72 * MB);  // 16 MB  -> 88 MB total

    // 1) x -> bf16
    {
        const int n4 = MTOT * D_MODEL / 4;  // 2,097,152
        cvt4_kernel<<<dim3(n4 / 256), 256, 0, stream>>>((const float4*)x, (v4bf*)xbf, n4);
    }
    // 2) weight transposes -> bf16
    {
        const int blocks = D_MODEL * D_MODEL / 256;  // 4096
        transpose_w_kernel<<<dim3(blocks), 256, 0, stream>>>(Wq, wqT);
        transpose_w_kernel<<<dim3(blocks), 256, 0, stream>>>(Wk, wkT);
        transpose_w_kernel<<<dim3(blocks), 256, 0, stream>>>(Wv, wvT);
        transpose_w_kernel<<<dim3(blocks), 256, 0, stream>>>(Wo, woT);
    }
    // 3) QKV projections
    const dim3 ggrid(MTOT / 128, D_MODEL / 128);     // 64 x 8
    gemm_bf16_kernel<1><<<ggrid, 256, 0, stream>>>(xbf, wqT, (void*)Qb,  nullptr);
    gemm_bf16_kernel<1><<<ggrid, 256, 0, stream>>>(xbf, wkT, (void*)Kbb, nullptr);
    gemm_bf16_kernel<2><<<ggrid, 256, 0, stream>>>(xbf, wvT, (void*)Vt,  nullptr);
    // 4) causal flash attention
    attn_kernel<<<dim3(SEQ / 16, NHEADS, BATCH), 32, 0, stream>>>(Qb, Kbb, Vt, Cb);
    // 5) output projection + bias (fp32 out)
    gemm_bf16_kernel<0><<<ggrid, 256, 0, stream>>>(Cb, woT, (void*)out, bo);
}
